// InputAttention_55430847922844
// MI455X (gfx1250) — matrix operand order repaired and checked by reading, compiled, not run on hardware
//
#include <hip/hip_runtime.h>
#include <cstdint>

// ---------------------------------------------------------------------------
// InputAttention fused kernel set for gfx1250 (MI455X, wave32, WMMA)
//   pass 0: We1, We2 -> bf16 (one-time, L2-resident; kills in-loop cvt VALU)
//   hs   = [h|c] @ We1 + be1                          (WMMA bf16, f32 acc)
//   ux   = X^T @ We2  (per batch)  -- fused, never stored
//   e    = tanh(hs + ux + be2) @ v                    (epilogue in-register)
//   attn = softmax_n(e)                               (bv dropped: invariant)
// ---------------------------------------------------------------------------

typedef __attribute__((ext_vector_type(16))) __bf16         v16bf;
typedef __attribute__((ext_vector_type(16))) unsigned short v16us;
typedef __attribute__((ext_vector_type(8)))  float          v8f;

#define B_DIM 128
#define M_DIM 256
#define T_DIM 512
#define N_DIM 512

union BfCast { v16us u; v16bf b; };

__device__ __forceinline__ unsigned short f2bf(float f) {
  unsigned int u = __float_as_uint(f);
  u += 0x7fffu + ((u >> 16) & 1u);       // round-to-nearest-even
  return (unsigned short)(u >> 16);
}

// A-fragment scatter target (ISA 7.12.2, 16-bit A 16x32):
//   lane = M + 16*((K>>3)&1) ; half = (K&7) | ((K&16)>>1)
__device__ __forceinline__ void scatter_a(unsigned short (*lds)[16],
                                          int Mi, int K, unsigned short val) {
  const int al = Mi + ((K & 8) ? 16 : 0);
  const int h  = (K & 7) | ((K & 16) >> 1);
  lds[al][h] = val;
}

// Direct bf16 B-fragment load: lane l holds weight column (l&15); element s is
// row k = 8*(l>>4) + (s&7) + 16*(s>>3) (strided T_DIM). No VALU conversion.
__device__ __forceinline__ v16bf load_b_frag_bf(const unsigned short* __restrict__ wr) {
  BfCast bb;
  const int hb = ((threadIdx.x & 31) >> 4) << 3;
#pragma unroll
  for (int s = 0; s < 16; ++s)
    bb.u[s] = wr[(size_t)(((s & 8) << 1) + hb + (s & 7)) * T_DIM];
  return bb.b;
}

// ---------------------------------------------------------------------------
// Kernel 0: f32 -> bf16 weight conversion (one-time). 4 elems/thread.
// ---------------------------------------------------------------------------
__global__ void __launch_bounds__(256)
cvt_bf16_kernel(const float* __restrict__ src, unsigned short* __restrict__ dst) {
  const int i = (blockIdx.x * 256 + threadIdx.x) * 4;
  float4 w = *(const float4*)(src + i);
  unsigned int lo = (unsigned int)f2bf(w.x) | ((unsigned int)f2bf(w.y) << 16);
  unsigned int hi = (unsigned int)f2bf(w.z) | ((unsigned int)f2bf(w.w) << 16);
  uint2 p; p.x = lo; p.y = hi;
  *(uint2*)(dst + i) = p;                 // global_store_b64
}

// ---------------------------------------------------------------------------
// Kernel 1: hs[b,k] = concat(hidden,cell)[b,:] @ We1[:,k] + be1[k]
// grid = B/16 blocks of 256 threads; wave w owns k-range [64w, 64w+64)
// ---------------------------------------------------------------------------
__global__ void __launch_bounds__(256)
hs_gemm_kernel(const float* __restrict__ hidden, const float* __restrict__ cell,
               const unsigned short* __restrict__ We1bf,
               const float* __restrict__ be1, float* __restrict__ hs) {
  const int b0   = blockIdx.x * 16;
  const int tid  = threadIdx.x;
  const int wave = tid >> 5;
  const int lane = tid & 31;
  __shared__ alignas(32) unsigned short lds_a[2][32][16];

  v8f acc[4] = {};
  const int k0w = wave * 64;

  for (int t = 0; t < 2 * M_DIM; t += 32) {
    const int buf = (t >> 5) & 1;
#pragma unroll
    for (int i = 0; i < 2; ++i) {
      const int idx = tid + i * 256;           // 32(K) x 16(M) tile elements
      const int K   = idx >> 4;
      const int Mi  = idx & 15;
      const int j   = t + K;
      const float x = (j < M_DIM) ? hidden[(b0 + Mi) * M_DIM + j]
                                  : cell[(b0 + Mi) * M_DIM + (j - M_DIM)];
      scatter_a(lds_a[buf], Mi, K, f2bf(x));
    }
    __syncthreads();
    BfCast a;
    a.u = *(const v16us*)&lds_a[buf][lane][0];  // 2x ds_load_b128
    const unsigned short* wrow = We1bf + (size_t)t * T_DIM + k0w + (lane & 15);
#pragma unroll
    for (int kt = 0; kt < 4; ++kt) {
      v16bf bfrag = load_b_frag_bf(wrow + kt * 16);
      acc[kt] = __builtin_amdgcn_wmma_f32_16x16x32_bf16(
          false, a.b, false, bfrag, (short)0, acc[kt], false, false);
    }
  }

#pragma unroll
  for (int kt = 0; kt < 4; ++kt) {
    const int k = k0w + kt * 16 + (lane & 15);
    const float bias = be1[k];
#pragma unroll
    for (int r = 0; r < 8; ++r) {
      const int m = r + ((lane >> 4) << 3);     // C/D layout: M=r (+8 hi half)
      hs[(size_t)(b0 + m) * T_DIM + k] = acc[kt][r] + bias;
    }
  }
}

// ---------------------------------------------------------------------------
// Kernel 2: fused  e[b,n] = sum_k tanh(hs[b,k]+be2[k]+ (X^T We2)[n,k]) * v[k]
// grid = (B, N/16); block = 256 thr / 8 waves; wave w -> k in [64w,64w+64)
// ---------------------------------------------------------------------------
__global__ void __launch_bounds__(256)
attn_main_kernel(const float* __restrict__ X,
                 const unsigned short* __restrict__ We2bf,
                 const float* __restrict__ be2, const float* __restrict__ vvec,
                 const float* __restrict__ hs, float* __restrict__ e_out) {
  const int b    = blockIdx.x;
  const int n0   = blockIdx.y * 16;
  const int tid  = threadIdx.x;
  const int wave = tid >> 5;
  const int lane = tid & 31;

  __shared__ alignas(32) unsigned short lds_a[2][32][16]; // double-buffered A
  __shared__ float lds_e[16];
  if (tid < 16) lds_e[tid] = 0.f;

  v8f acc[4] = {};
  const int k0w = wave * 64;
  const float* __restrict__ Xb = X + (size_t)b * T_DIM * N_DIM;

  for (int t = 0; t < T_DIM; t += 32) {
    const int buf = (t >> 5) & 1;
    // stage transposed X tile: A[n][t'] = X[b, t+t', n0+n], f32 -> bf16,
    // scattered directly into per-lane WMMA A-fragment order.
#pragma unroll
    for (int i = 0; i < 2; ++i) {
      const int idx = tid + i * 256;
      const int K   = idx >> 4;
      const int Mi  = idx & 15;
      const float x = Xb[(size_t)(t + K) * N_DIM + n0 + Mi];
      if (t + 32 < T_DIM)  // pull next tile toward L2 (global_prefetch_b8)
        __builtin_prefetch(&Xb[(size_t)(t + 32 + K) * N_DIM + n0 + Mi], 0, 0);
      scatter_a(lds_a[buf], Mi, K, f2bf(x));
    }
    __syncthreads();
    BfCast a;
    a.u = *(const v16us*)&lds_a[buf][lane][0];
    const unsigned short* wrow = We2bf + (size_t)t * T_DIM + k0w + (lane & 15);
#pragma unroll
    for (int kt = 0; kt < 4; ++kt) {
      v16bf bfrag = load_b_frag_bf(wrow + kt * 16);   // L2-resident bf16 row
      acc[kt] = __builtin_amdgcn_wmma_f32_16x16x32_bf16(
          false, a.b, false, bfrag, (short)0, acc[kt], false, false);
    }
  }

  // Fused epilogue: tanh + dot with v, reduce over this wave's 64 k values.
  const float* __restrict__ hsrow = hs + (size_t)b * T_DIM;
#pragma unroll
  for (int kt = 0; kt < 4; ++kt) {
    const int k    = k0w + kt * 16 + (lane & 15);    // C/D layout: N = lane&15
    const float h  = hsrow[k] + be2[k];
    const float vk = vvec[k];
#pragma unroll
    for (int r = 0; r < 8; ++r) {
      float s = tanhf(acc[kt][r] + h) * vk;
      // sum the 16 k-lanes of this half-wave
      s += __shfl_xor(s, 1, 32);
      s += __shfl_xor(s, 2, 32);
      s += __shfl_xor(s, 4, 32);
      s += __shfl_xor(s, 8, 32);
      if ((lane & 15) == 0) {
        const int n_loc = r + ((lane >> 4) << 3);
        atomicAdd(&lds_e[n_loc], s);                 // cross-wave (k) reduce
      }
    }
  }
  __syncthreads();
  if (tid < 16) e_out[(size_t)b * N_DIM + n0 + tid] = lds_e[tid];
}

// ---------------------------------------------------------------------------
// Kernel 3: row softmax over n (512) per batch. 256 thr, 2 elems/thread.
// ---------------------------------------------------------------------------
__global__ void __launch_bounds__(256)
softmax_kernel(const float* __restrict__ e, float* __restrict__ out) {
  const int b    = blockIdx.x;
  const int tid  = threadIdx.x;
  const int wave = tid >> 5;
  const int lane = tid & 31;
  __shared__ float redm[8];
  __shared__ float reds[8];

  const float x0 = e[(size_t)b * N_DIM + tid];
  const float x1 = e[(size_t)b * N_DIM + 256 + tid];

  float m = fmaxf(x0, x1);
#pragma unroll
  for (int sh = 16; sh > 0; sh >>= 1) m = fmaxf(m, __shfl_xor(m, sh, 32));
  if (lane == 0) redm[wave] = m;
  __syncthreads();
  float M = redm[0];
#pragma unroll
  for (int i = 1; i < 8; ++i) M = fmaxf(M, redm[i]);

  const float ex0 = __expf(x0 - M);
  const float ex1 = __expf(x1 - M);
  float s = ex0 + ex1;
#pragma unroll
  for (int sh = 16; sh > 0; sh >>= 1) s += __shfl_xor(s, sh, 32);
  if (lane == 0) reds[wave] = s;
  __syncthreads();
  float S = 0.f;
#pragma unroll
  for (int i = 0; i < 8; ++i) S += reds[i];
  const float inv = 1.0f / S;

  out[(size_t)b * N_DIM + tid]       = ex0 * inv;
  out[(size_t)b * N_DIM + 256 + tid] = ex1 * inv;
}

// ---------------------------------------------------------------------------
extern "C" void kernel_launch(void* const* d_in, const int* in_sizes, int n_in,
                              void* d_out, int out_size, void* d_ws, size_t ws_size,
                              hipStream_t stream) {
  const float* hidden = (const float*)d_in[0];
  const float* cell   = (const float*)d_in[1];
  const float* X      = (const float*)d_in[2];
  const float* We1    = (const float*)d_in[3];
  const float* be1    = (const float*)d_in[4];
  const float* We2    = (const float*)d_in[5];
  const float* be2    = (const float*)d_in[6];
  const float* vvec   = (const float*)d_in[7];
  // d_in[8] = bv : softmax is invariant to a constant shift -> dropped.
  float* out = (float*)d_out;

  // Workspace layout (1.5 MB total)
  float*          hs    = (float*)d_ws;                       // [B,T] 256 KB
  float*          ee    = hs + (size_t)B_DIM * T_DIM;         // [B,N] 256 KB
  unsigned short* We1bf = (unsigned short*)(ee + (size_t)B_DIM * N_DIM); // 512 KB
  unsigned short* We2bf = We1bf + (size_t)2 * M_DIM * T_DIM;             // 512 KB

  const int nW = (2 * M_DIM * T_DIM) / (256 * 4);             // 256 blocks
  hipLaunchKernelGGL(cvt_bf16_kernel, dim3(nW), dim3(256), 0, stream, We1, We1bf);
  hipLaunchKernelGGL(cvt_bf16_kernel, dim3(nW), dim3(256), 0, stream, We2, We2bf);

  hipLaunchKernelGGL(hs_gemm_kernel, dim3(B_DIM / 16), dim3(256), 0, stream,
                     hidden, cell, We1bf, be1, hs);
  hipLaunchKernelGGL(attn_main_kernel, dim3(B_DIM, N_DIM / 16), dim3(256), 0,
                     stream, X, We2bf, be2, vvec, hs, ee);
  hipLaunchKernelGGL(softmax_kernel, dim3(B_DIM), dim3(256), 0, stream, ee, out);
}
